// SpatialAttention_58626303590953
// MI455X (gfx1250) — matrix operand order repaired and checked by reading, not compiled, not run on hardware
//
#include <hip/hip_runtime.h>
#include <hip/hip_bf16.h>
#include <stdint.h>

// ---------------------------------------------------------------- constants
#define WSZ   8
#define NH    16
#define DIM   512
#define HD    32              // head dim
#define NTOK  64              // tokens per window
#define BV    4
#define HV    128
#define WV    128
#define NWIN_PER_B 256        // 16 x 16 windows
#define NWIN  (BV * NWIN_PER_B)   // 1024
#define RELN  ((2*WSZ-1)*(2*WSZ-1))  // 225

// ---------------------------------------------------------------- types
typedef __attribute__((ext_vector_type(16))) __bf16         v16bf;
typedef __attribute__((ext_vector_type(8)))  float          v8f;
typedef __attribute__((ext_vector_type(8)))  unsigned short us8;
typedef __attribute__((ext_vector_type(4)))  unsigned int   v4u;
typedef __attribute__((ext_vector_type(8)))  int            v8i;
typedef __attribute__((ext_vector_type(4)))  int            v4i;

#if defined(__has_builtin)
#if __has_builtin(__builtin_amdgcn_tensor_load_to_lds) && \
    __has_builtin(__builtin_amdgcn_s_wait_tensorcnt)
#define HAVE_TDM 1
#endif
#endif
#ifndef HAVE_TDM
#define HAVE_TDM 0
#endif

static __device__ __forceinline__ v8f vzero8() {
  v8f z = {0.f, 0.f, 0.f, 0.f, 0.f, 0.f, 0.f, 0.f};
  return z;
}

static __device__ __forceinline__ unsigned short f2bf(float f) {
  unsigned int u = __float_as_uint(f);
  u += 0x7fffu + ((u >> 16) & 1u);          // round-to-nearest-even
  return (unsigned short)(u >> 16);
}

static __device__ __forceinline__ unsigned lds_addr_of(const void* p) {
  // generic->LDS: hardware uses addr[31:0] for the LDS aperture
  return (unsigned)(uintptr_t)p;
}

union FragBF { v16bf v; us8 h[2]; };

// A fragment (16x32 bf16): row-major src, row stride rs (elements).
static __device__ __forceinline__ v16bf frag_a(const unsigned short* src,
                                               int rs, int r0, int kb, int lane) {
  const int row = r0 + (lane & 15);
  const int hlf = lane >> 4;
  const unsigned short* p = src + row * rs + kb + hlf * 8;
  FragBF f;
  f.h[0] = *(const us8*)p;
  f.h[1] = *(const us8*)(p + 16);
  return f.v;
}

// B fragment (32x16 bf16): element (k,n) stored at src[(c0+n)*cs + kb + k].
static __device__ __forceinline__ v16bf frag_b(const unsigned short* src,
                                               int cs, int c0, int kb, int lane) {
  const int col = c0 + (lane & 15);
  const int hlf = lane >> 4;
  const unsigned short* p = src + col * cs + kb + hlf * 8;
  FragBF f;
  f.h[0] = *(const us8*)p;
  f.h[1] = *(const us8*)(p + 16);
  return f.v;
}

static __device__ __forceinline__ v8f wmma_bf16(v16bf a, v16bf b, v8f c) {
  return __builtin_amdgcn_wmma_f32_16x16x32_bf16(
      false, a, false, b, (short)0, c, false, false);
}

#if HAVE_TDM
// TDM 2-D tile load: memory -> LDS, bf16 elements (data_size=1 -> 2 bytes).
// D# per CDNA5 ISA: group0 = {count|flags, lds_addr,
// global_addr lo, global_addr hi | type=2}, group1 packs data_size, tensor
// dims, tile dims, dim0 stride.  Remaining groups zero (<=2D tensor).
// This toolchain's builtin takes 6 args: (v4u, v8i, v4i, v4i, v8i, i32 cpol).
static __device__ __forceinline__ void tdm_load_2d_bf16(
    unsigned lds_byte_addr, const void* gptr,
    unsigned tile_d0, unsigned tile_d1,
    unsigned tensor_d0, unsigned tensor_d1, unsigned d0_stride) {
  unsigned long long ga = (unsigned long long)(uintptr_t)gptr;
  v4u g0;
  g0.x = 1u;                                        // count=1 (user descriptor)
  g0.y = lds_byte_addr;                             // lds_addr [63:32]
  g0.z = (unsigned)(ga & 0xffffffffu);              // global_addr lo
  g0.w = (unsigned)((ga >> 32) & 0x01ffffffu) | (2u << 30);  // hi | type=2
  v8i g1;
  g1[0] = (int)(1u << 16);                          // data_size=1 (2B), mask=0
  g1[1] = (int)((tensor_d0 & 0xffffu) << 16);       // tensor_dim0[15:0]
  g1[2] = (int)(((tensor_d0 >> 16) & 0xffffu) |
                ((tensor_d1 & 0xffffu) << 16));     // dim0 hi | tensor_dim1 lo
  g1[3] = (int)(((tensor_d1 >> 16) & 0xffffu) |
                ((tile_d0 & 0xffffu) << 16));       // dim1 hi | tile_dim0
  g1[4] = (int)(tile_d1 & 0xffffu);                 // tile_dim1, tile_dim2=0
  g1[5] = (int)d0_stride;                           // tensor_dim0_stride lo32
  g1[6] = 0;                                        // stride hi | dim1 stride
  g1[7] = 0;
  v4i z4 = {0, 0, 0, 0};
  v8i z8 = {0, 0, 0, 0, 0, 0, 0, 0};
  __builtin_amdgcn_tensor_load_to_lds(g0, g1, z4, z4, z8, 0);
}
#endif

// ================================================================ kernel 1:
// windowize + LayerNorm + fp32->bf16.  grid = B*H*(W/64), block = 256.
__global__ void __launch_bounds__(256)
k_ln_window(const float* __restrict__ x, const float* __restrict__ gamma,
            const float* __restrict__ beta, unsigned short* __restrict__ xn) {
  const int tid = threadIdx.x;
  const int tx  = tid & 63;
  const int ty  = tid >> 6;          // 0..3 -> owns 128 channels
  const int bid = blockIdx.x;
  const int wchunk = bid & 1;        // W/64 == 2
  const int h = (bid >> 1) & (HV - 1);
  const int b = bid >> 8;
  const int w = wchunk * 64 + tx;
  const size_t plane = (size_t)HV * WV;
  const float* px = x + (size_t)b * DIM * plane + (size_t)h * WV + w;

  __shared__ float s_sum[4][64];
  __shared__ float s_sq[4][64];
  float s = 0.f, s2 = 0.f;
  for (int c = ty * 128; c < ty * 128 + 128; ++c) {
    float v = px[(size_t)c * plane];       // coalesced across tx
    s += v; s2 += v * v;
  }
  s_sum[ty][tx] = s; s_sq[ty][tx] = s2;
  __syncthreads();

  __shared__ float s_mu[64], s_rs[64];
  if (tid < 64) {
    float S = s_sum[0][tid] + s_sum[1][tid] + s_sum[2][tid] + s_sum[3][tid];
    float Q = s_sq[0][tid] + s_sq[1][tid] + s_sq[2][tid] + s_sq[3][tid];
    float mu  = S * (1.f / DIM);
    float var = Q * (1.f / DIM) - mu * mu;
    s_mu[tid] = mu;
    s_rs[tid] = rsqrtf(var + 1e-5f);
  }
  __syncthreads();

  const float mu = s_mu[tx], rs = s_rs[tx];
  const int wy = h >> 3, tyy = h & 7;
  const int wx = w >> 3, txx = w & 7;
  const int win = b * NWIN_PER_B + wy * 16 + wx;
  const int n   = tyy * WSZ + txx;
  unsigned short* po = xn + (size_t)(win * NTOK + n) * DIM;
  for (int c = ty * 128; c < ty * 128 + 128; ++c) {
    float v = px[(size_t)c * plane];
    po[c] = f2bf((v - mu) * rs * gamma[c] + beta[c]);
  }
}

// ================================================================ kernel 2:
__global__ void k_cvt_bf16(const float* __restrict__ a,
                           unsigned short* __restrict__ o, int n) {
  for (int i = blockIdx.x * blockDim.x + threadIdx.x; i < n;
       i += gridDim.x * blockDim.x)
    o[i] = f2bf(a[i]);
}

// ================================================================ kernel 3:
// fused per-(window, head) attention.  grid = (1024, 16), block = 128 (4 waves).
__global__ void __launch_bounds__(128)
k_window_attn(const unsigned short* __restrict__ xn,    // (NWIN,64,512) bf16
              const unsigned short* __restrict__ wqkv,  // (512,1536) bf16
              const float* __restrict__ bias_table,     // (225,16) fp32
              unsigned short* __restrict__ obuf) {      // (NWIN,64,512) bf16
  const int win  = blockIdx.x;
  const int head = blockIdx.y;
  const int tid  = threadIdx.x;
  const int lane = tid & 31;
  const int wave = tid >> 5;
  const int hlf  = lane >> 4;
  const int colL = lane & 15;

  __shared__ __align__(16) unsigned short Xc[2][NTOK * 32]; // dbl-buf x chunk
  __shared__ __align__(16) unsigned short Wt[2][96 * 32];   // dbl-buf [col][k]
  __shared__ __align__(16) unsigned short Qs[NTOK * HD];    // [n][d]
  __shared__ __align__(16) unsigned short Ks[NTOK * HD];    // [m][d]
  __shared__ __align__(16) unsigned short Vt[HD * NTOK];    // [d][m]
  __shared__ __align__(16) float          Ss[NTOK * NTOK];
  __shared__ __align__(16) unsigned short Ps[NTOK * NTOK];
  __shared__ float sbias[RELN];

  for (int i = tid; i < RELN; i += 128) sbias[i] = bias_table[i * NH + head];

  const unsigned short* xw = xn + (size_t)win * NTOK * DIM;

  // producer helpers -------------------------------------------------------
  auto stage_w = [&](int p, int kc) {
    for (int idx = tid; idx < 96 * 32; idx += 128) {
      int kk = idx / 96, col = idx - kk * 96;
      int s = col >> 5, j = col & 31;
      Wt[p][col * 32 + kk] =
          wqkv[(size_t)(kc * 32 + kk) * (3 * DIM) + s * DIM + head * HD + j];
    }
    if (kc + 1 < DIM / 32)
      __builtin_prefetch(&wqkv[(size_t)((kc + 1) * 32 + (tid & 31)) *
                               (3 * DIM) + head * HD], 0, 1);
  };
#if HAVE_TDM
  // each wave DMAs its own 16x32 subtile (one TENSOR_LOAD_TO_LDS per wave)
  auto stage_x = [&](int p, int kc) {
    tdm_load_2d_bf16(lds_addr_of(&Xc[p][wave * 16 * 32]),
                     xw + (size_t)(wave * 16) * DIM + kc * 32,
                     /*tile*/ 32, 16, /*tensor*/ DIM, NTOK, /*stride*/ DIM);
  };
#else
  auto stage_x = [&](int p, int kc) {
    for (int idx = tid; idx < 256; idx += 128) {
      int row = idx >> 2, off = (idx & 3) * 8;
      *(us8*)&Xc[p][row * 32 + off] =
          *(const us8*)&xw[row * DIM + kc * 32 + off];
    }
  };
#endif

  // ---- stage A: QKV = Xn(64x512) @ Wqkv[:, head cols] (512x96) -------------
  v8f acc[6];
#pragma unroll
  for (int t = 0; t < 6; ++t) acc[t] = vzero8();

  stage_x(0, 0);
  stage_w(0, 0);
  for (int kc = 0; kc < DIM / 32; ++kc) {
    const int p = kc & 1;
    if (kc + 1 < DIM / 32) {
      stage_x(p ^ 1, kc + 1);          // async into the other buffer
      stage_w(p ^ 1, kc + 1);
#if HAVE_TDM
      __builtin_amdgcn_s_wait_tensorcnt(1);   // chunk kc's DMA done
#endif
    } else {
#if HAVE_TDM
      __builtin_amdgcn_s_wait_tensorcnt(0);
#endif
    }
    __syncthreads();
    v16bf a = frag_a(Xc[p], 32, wave * 16, 0, lane);
#pragma unroll
    for (int nt = 0; nt < 6; ++nt) {
      v16bf bf = frag_b(Wt[p], 32, nt * 16, 0, lane);
      acc[nt] = wmma_bf16(a, bf, acc[nt]);
    }
    __syncthreads();
  }

  // write Q (pre-scaled), K row-major; V transposed
  const float scale = 0.17677669529663687f;   // 32^-0.5
#pragma unroll
  for (int v = 0; v < 8; ++v) {
    int m = wave * 16 + v + 8 * hlf;
    Qs[m * HD + colL]       = f2bf(acc[0][v] * scale);
    Qs[m * HD + 16 + colL]  = f2bf(acc[1][v] * scale);
    Ks[m * HD + colL]       = f2bf(acc[2][v]);
    Ks[m * HD + 16 + colL]  = f2bf(acc[3][v]);
    Vt[colL * NTOK + m]        = f2bf(acc[4][v]);
    Vt[(16 + colL) * NTOK + m] = f2bf(acc[5][v]);
  }
  __syncthreads();

  // ---- stage B: S = Q @ K^T + rel_bias -------------------------------------
  {
    v16bf aq = frag_a(Qs, HD, wave * 16, 0, lane);
#pragma unroll
    for (int ct = 0; ct < 4; ++ct) {
      v8f sc = vzero8();
      v16bf bk = frag_b(Ks, HD, ct * 16, 0, lane);
      sc = wmma_bf16(aq, bk, sc);
#pragma unroll
      for (int v = 0; v < 8; ++v) {
        int n = wave * 16 + v + 8 * hlf;
        int m = ct * 16 + colL;
        int dy = (n >> 3) - (m >> 3) + (WSZ - 1);
        int dx = (n & 7) - (m & 7) + (WSZ - 1);
        Ss[n * NTOK + m] = sc[v] + sbias[dy * (2 * WSZ - 1) + dx];
      }
    }
  }
  __syncthreads();

  // ---- softmax (one thread per row, fp32) ----------------------------------
  if (tid < NTOK) {
    const float* row = &Ss[tid * NTOK];
    float mx = -3.4e38f;
    for (int m = 0; m < NTOK; ++m) mx = fmaxf(mx, row[m]);
    float sum = 0.f;
    for (int m = 0; m < NTOK; ++m) sum += __expf(row[m] - mx);
    float inv = 1.f / sum;
    unsigned short* prow = &Ps[tid * NTOK];
    for (int m = 0; m < NTOK; ++m) prow[m] = f2bf(__expf(row[m] - mx) * inv);
  }
  __syncthreads();

  // ---- stage C: O = P(64x64) @ V(64x32) ------------------------------------
  v8f oacc0 = vzero8(), oacc1 = vzero8();
#pragma unroll
  for (int ks = 0; ks < 2; ++ks) {
    v16bf ap = frag_a(Ps, NTOK, wave * 16, ks * 32, lane);
    v16bf b0 = frag_b(Vt, NTOK, 0,  ks * 32, lane);
    v16bf b1 = frag_b(Vt, NTOK, 16, ks * 32, lane);
    oacc0 = wmma_bf16(ap, b0, oacc0);
    oacc1 = wmma_bf16(ap, b1, oacc1);
  }
  unsigned short* ow = obuf + (size_t)win * NTOK * DIM + head * HD;
#pragma unroll
  for (int v = 0; v < 8; ++v) {
    int m = wave * 16 + v + 8 * hlf;
    ow[(size_t)m * DIM + colL]      = f2bf(oacc0[v]);
    ow[(size_t)m * DIM + 16 + colL] = f2bf(oacc1[v]);
  }
}

// ================================================================ kernel 4:
// proj GEMM + bias + un-windowize to NCHW fp32.  grid = (1024, 4), block 128.
__global__ void __launch_bounds__(128)
k_proj(const unsigned short* __restrict__ obuf,   // (NWIN,64,512) bf16
       const unsigned short* __restrict__ wproj,  // (512,512) bf16
       const float* __restrict__ bproj,
       float* __restrict__ out) {
  const int win = blockIdx.x;
  const int cg  = blockIdx.y;            // cols cg*128..+127
  const int tid = threadIdx.x;
  const int lane = tid & 31;
  const int wave = tid >> 5;
  const int hlf  = lane >> 4;
  const int colL = lane & 15;
  const int c0 = cg * 128;

  __shared__ __align__(16) unsigned short Oc[2][NTOK * 32];
  __shared__ __align__(16) unsigned short Wpt[2][128 * 32];

  const unsigned short* ow = obuf + (size_t)win * NTOK * DIM;

  auto stage_w = [&](int p, int kc) {
    for (int idx = tid; idx < 128 * 32; idx += 128) {
      int kk = idx >> 7, col = idx & 127;
      Wpt[p][col * 32 + kk] = wproj[(size_t)(kc * 32 + kk) * DIM + c0 + col];
    }
    if (kc + 1 < DIM / 32)
      __builtin_prefetch(&wproj[(size_t)((kc + 1) * 32 + (tid & 31)) * DIM + c0],
                         0, 1);
  };
#if HAVE_TDM
  auto stage_o = [&](int p, int kc) {
    tdm_load_2d_bf16(lds_addr_of(&Oc[p][wave * 16 * 32]),
                     ow + (size_t)(wave * 16) * DIM + kc * 32,
                     /*tile*/ 32, 16, /*tensor*/ DIM, NTOK, /*stride*/ DIM);
  };
#else
  auto stage_o = [&](int p, int kc) {
    for (int idx = tid; idx < 256; idx += 128) {
      int row = idx >> 2, off = (idx & 3) * 8;
      *(us8*)&Oc[p][row * 32 + off] =
          *(const us8*)&ow[row * DIM + kc * 32 + off];
    }
  };
#endif

  v8f acc[8];
#pragma unroll
  for (int t = 0; t < 8; ++t) acc[t] = vzero8();

  stage_o(0, 0);
  stage_w(0, 0);
  for (int kc = 0; kc < DIM / 32; ++kc) {
    const int p = kc & 1;
    if (kc + 1 < DIM / 32) {
      stage_o(p ^ 1, kc + 1);
      stage_w(p ^ 1, kc + 1);
#if HAVE_TDM
      __builtin_amdgcn_s_wait_tensorcnt(1);
#endif
    } else {
#if HAVE_TDM
      __builtin_amdgcn_s_wait_tensorcnt(0);
#endif
    }
    __syncthreads();
    v16bf a = frag_a(Oc[p], 32, wave * 16, 0, lane);
#pragma unroll
    for (int nt = 0; nt < 8; ++nt) {
      v16bf bf = frag_b(Wpt[p], 32, nt * 16, 0, lane);
      acc[nt] = wmma_bf16(a, bf, acc[nt]);
    }
    __syncthreads();
  }

  const int b  = win >> 8;
  const int wy = (win >> 4) & 15;
  const int wx = win & 15;
#pragma unroll
  for (int nt = 0; nt < 8; ++nt) {
    int c = c0 + nt * 16 + colL;
    float bias = bproj[c];
#pragma unroll
    for (int v = 0; v < 8; ++v) {
      int n = wave * 16 + v + 8 * hlf;
      int h = wy * WSZ + (n >> 3);
      int w = wx * WSZ + (n & 7);
      out[(((size_t)b * DIM + c) * HV + h) * WV + w] = acc[nt][v] + bias;
    }
  }
}

// ================================================================ launcher
extern "C" void kernel_launch(void* const* d_in, const int* in_sizes, int n_in,
                              void* d_out, int out_size, void* d_ws,
                              size_t ws_size, hipStream_t stream) {
  const float* x     = (const float*)d_in[0];
  const float* gamma = (const float*)d_in[1];
  const float* beta  = (const float*)d_in[2];
  const float* wqkv  = (const float*)d_in[3];
  const float* wproj = (const float*)d_in[4];
  const float* bproj = (const float*)d_in[5];
  const float* btab  = (const float*)d_in[6];
  float* out = (float*)d_out;

  // workspace: xn (64MB) | obuf (64MB) | wqkv bf16 (1.5MB) | wproj bf16 (0.5MB)
  unsigned short* ws   = (unsigned short*)d_ws;
  unsigned short* xn   = ws;
  unsigned short* obuf = xn   + (size_t)NWIN * NTOK * DIM;
  unsigned short* wqb  = obuf + (size_t)NWIN * NTOK * DIM;
  unsigned short* wpb  = wqb  + (size_t)DIM * 3 * DIM;

  k_ln_window<<<dim3(BV * HV * (WV / 64)), 256, 0, stream>>>(x, gamma, beta, xn);
  k_cvt_bf16<<<dim3(512), 256, 0, stream>>>(wqkv, wqb, DIM * 3 * DIM);
  k_cvt_bf16<<<dim3(256), 256, 0, stream>>>(wproj, wpb, DIM * DIM);
  k_window_attn<<<dim3(NWIN, NH), 128, 0, stream>>>(xn, wqb, btab, obuf);
  k_proj<<<dim3(NWIN, 4), 128, 0, stream>>>(obuf, wpb, bproj, out);
}
